// FreeConvNetwork_2619930051755
// MI455X (gfx1250) — matrix operand order repaired and checked by reading, not compiled, not run on hardware
//
#include <hip/hip_runtime.h>
#include <hip/hip_bf16.h>

typedef __attribute__((ext_vector_type(16))) _Float16 v16h;
typedef __attribute__((ext_vector_type(8)))  float    v8f;
typedef int v4i_ __attribute__((vector_size(16)));

#define NB 64
// L1: in (64,3,218,178) stride2 k6 -> (64,107,87)
// L2: -> (64,102,82)   L3: -> (64,97,77)   L4: -> (64,92,72) -> pool -> (64,46,36)=1656
#define FC1_K 1656
#define FC1_KP 1664
#define FC1_N 300
#define FC1_NP 304
#define FC2_K 300
#define FC2_KP 320
#define FC2_N 200
#define FC2_NP 208

#define AS1 __attribute__((address_space(1)))
#define AS3 __attribute__((address_space(3)))

#if defined(__has_builtin)
#if __has_builtin(__builtin_amdgcn_global_load_async_to_lds_b128) && \
    __has_builtin(__builtin_amdgcn_s_wait_asynccnt)
#define ASYNC_LDS 1
#endif
#endif

// Stage 16 bytes global -> LDS. Async (ASYNCcnt-tracked) when available.
__device__ __forceinline__ void stage16(const float* g, float* l) {
#ifdef ASYNC_LDS
    __builtin_amdgcn_global_load_async_to_lds_b128(
        (AS1 v4i_*)(void*)g, (AS3 v4i_*)(void*)l, 0, 0);
#else
    float4 t = *(const float4*)g;
    *(float4*)l = t;
#endif
}

__device__ __forceinline__ void stage_wait() {
#ifdef ASYNC_LDS
    __builtin_amdgcn_s_wait_asynccnt(0);
#endif
    __syncthreads();
}

// ------------- conv layer 1 (3ch, stride 2), weights staged in LDS -------------
// Block: 64 output positions x all 64 batches. tid%64 = local pos, tid/64 = batch group.
__global__ void conv1_staged_k(const float* __restrict__ x, const float* __restrict__ w,
                               const float* __restrict__ bias, float* __restrict__ out) {
    const int OH = 107, OW = 87, TOT = OH * OW;
    __shared__ __align__(16) float lw[3 * 64 * 36];
    __shared__ float lb[64];
    int tid = threadIdx.x;
    int p0 = blockIdx.x * 64;
    int npos = TOT - p0; if (npos > 64) npos = 64;

    // stage weights: 3 channel runs of npos*36 contiguous floats each (16B chunks)
    int chunks = npos * 9; // 9 x 16B per position per channel
    for (int t = tid; t < 3 * chunks; t += blockDim.x) {
        int ch = t / chunks, c = t % chunks;
        stage16(w + ((size_t)ch * TOT + p0) * 36 + c * 4, lw + ch * 2304 + c * 4);
    }
    if (tid < npos) lb[tid] = bias[p0 + tid];
    stage_wait();

    int pl = tid & 63;
    int g  = tid >> 6;          // batch group 0..3
    int p  = p0 + pl;
    if (pl < npos) {
        int ox = p / OW, oy = p % OW;
        for (int b = g * 16; b < g * 16 + 16; ++b) {
            float s = 0.f;
#pragma unroll
            for (int ch = 0; ch < 3; ++ch) {
                const float* xp = x + (((size_t)b * 3 + ch) * 218 + ox * 2) * 178 + oy * 2;
                const float* wp = lw + ch * 2304 + pl * 36;
                __builtin_prefetch(xp + 2 * 178, 0, 0);  // global_prefetch_b8
#pragma unroll
                for (int i = 0; i < 6; ++i)
#pragma unroll
                    for (int j = 0; j < 6; ++j)
                        s += xp[i * 178 + j] * wp[i * 6 + j];
            }
            s += lb[pl];
            out[((size_t)b * OH + ox) * OW + oy] = s > 0.f ? s : 0.f;
        }
    }
}

// ------------- conv layers 2/3/4 (1ch, stride 1), weights staged in LDS -------------
__global__ void convm_staged_k(const float* __restrict__ in, const float* __restrict__ w,
                               const float* __restrict__ bias, float* __restrict__ out,
                               int IH, int IW, int OH, int OW) {
    __shared__ __align__(16) float lw[64 * 36];
    __shared__ float lb[64];
    int tid = threadIdx.x;
    int TOT = OH * OW;
    int p0 = blockIdx.x * 64;
    int npos = TOT - p0; if (npos > 64) npos = 64;

    int chunks = npos * 9;
    for (int c = tid; c < chunks; c += blockDim.x)
        stage16(w + (size_t)p0 * 36 + c * 4, lw + c * 4);
    if (tid < npos) lb[tid] = bias[p0 + tid];
    stage_wait();

    int pl = tid & 63;
    int g  = tid >> 6;
    int p  = p0 + pl;
    if (pl < npos) {
        int ox = p / OW, oy = p % OW;
        const float* wp = lw + pl * 36;
        for (int b = g * 16; b < g * 16 + 16; ++b) {
            const float* ip = in + ((size_t)b * IH + ox) * IW + oy;
            float s = 0.f;
#pragma unroll
            for (int i = 0; i < 6; ++i)
#pragma unroll
                for (int j = 0; j < 6; ++j)
                    s += ip[i * IW + j] * wp[i * 6 + j];
            s += lb[pl];
            out[((size_t)b * OH + ox) * OW + oy] = s > 0.f ? s : 0.f;
        }
    }
}

// ------------- 2x2 maxpool (64,92,72) -> flatten f16 (64,1656 in 1664-pad rows) -------------
__global__ void pool_k(const float* __restrict__ a4, _Float16* __restrict__ h16) {
    int idx = blockIdx.x * blockDim.x + threadIdx.x;
    if (idx >= NB * 46 * 36) return;
    int py = idx % 36; int t = idx / 36; int px = t % 46; int bb = t / 46;
    const float* p = a4 + ((size_t)bb * 92 + 2 * px) * 72 + 2 * py;
    float m = p[0];
    if (p[1]  > m) m = p[1];
    if (p[72] > m) m = p[72];
    if (p[73] > m) m = p[73];
    h16[(size_t)bb * FC1_KP + px * 36 + py] = (_Float16)m;
}

// ---------------- helpers: zero fill / convert+pad f32 -> f16 ----------------
__global__ void fill0_k(_Float16* __restrict__ p, int n) {
    int i = blockIdx.x * blockDim.x + threadIdx.x;
    if (i < n) p[i] = (_Float16)0.f;
}

__global__ void cvtpad_k(const float* __restrict__ src, _Float16* __restrict__ dst,
                         int R, int C, int Rp, int Cp) {
    int i = blockIdx.x * blockDim.x + threadIdx.x;
    if (i >= Rp * Cp) return;
    int c = i % Cp, r = i / Cp;
    float v = (r < R && c < C) ? src[(size_t)r * C + c] : 0.f;
    dst[i] = (_Float16)v;
}

// ---------------- WMMA FC: D(16x16 f32) = A(16x32 f16) x B(32x16 f16) + C ----------------
// One wave per 16x16 tile; EXEC stays all-ones (no divergent exits before WMMA).
__global__ void wmma_fc_k(const _Float16* __restrict__ A, const _Float16* __restrict__ Bw,
                          const float* __restrict__ bias, int Kpad, int Nact,
                          float* __restrict__ out32, int os32,
                          _Float16* __restrict__ out16, int os16, int relu) {
    int lane  = threadIdx.x;         // 0..31 (wave32)
    int l15   = lane & 15;
    int agrp  = (lane >> 4) * 16;    // A-fragment K group offset
    int bgrp  = (lane >> 4) * 16;    // B-fragment K group offset
    int ntile = blockIdx.x, mtile = blockIdx.y;

    const _Float16* arow = A  + (size_t)(mtile * 16 + l15) * Kpad;
    const _Float16* brow = Bw + (size_t)(ntile * 16 + l15) * Kpad;

    v8f acc = {};
    for (int kb = 0; kb < Kpad; kb += 32) {
        v16h av, bv;
#pragma unroll
        for (int h = 0; h < 16; ++h) {
            int ka = kb + agrp + h;                     // contiguous K order, matches B
            av[h] = arow[ka];
        }
#pragma unroll
        for (int h = 0; h < 16; ++h)                    // contiguous K order, matches A
            bv[h] = brow[kb + bgrp + h];
        acc = __builtin_amdgcn_wmma_f32_16x16x32_f16(
            /*neg_a=*/false, av, /*neg_b=*/false, bv,
            /*c_mod=*/(short)0, acc, /*reuse_a=*/false, /*reuse_b=*/false);
    }

    int n = ntile * 16 + l15;
    int mbase = mtile * 16 + ((lane >> 4) * 8);         // C/D layout: M = r + (lane>=16 ? 8 : 0)
#pragma unroll
    for (int r = 0; r < 8; ++r) {
        int m = mbase + r;
        if (n < Nact) {
            float v = acc[r] + bias[n];
            if (relu && v < 0.f) v = 0.f;
            if (out32) out32[(size_t)m * os32 + n] = v;
            if (out16) out16[(size_t)m * os16 + n] = (_Float16)v;
        }
    }
}

extern "C" void kernel_launch(void* const* d_in, const int* in_sizes, int n_in,
                              void* d_out, int out_size, void* d_ws, size_t ws_size,
                              hipStream_t stream) {
    const float* x     = (const float*)d_in[0];
    const float* w1    = (const float*)d_in[1];
    const float* b1    = (const float*)d_in[2];
    const float* w2    = (const float*)d_in[3];
    const float* b2    = (const float*)d_in[4];
    const float* w3    = (const float*)d_in[5];
    const float* b3    = (const float*)d_in[6];
    const float* w4    = (const float*)d_in[7];
    const float* b4    = (const float*)d_in[8];
    const float* fc1_w = (const float*)d_in[9];
    const float* fc1_b = (const float*)d_in[10];
    const float* fc2_w = (const float*)d_in[11];
    const float* fc2_b = (const float*)d_in[12];
    float* out = (float*)d_out;

    // workspace carve-out (256B aligned)
    char* ws = (char*)d_ws;
    size_t off = 0;
    auto carve = [&](size_t bytes) { char* p = ws + off; off = (off + bytes + 255) & ~(size_t)255; return p; };
    float*    a1  = (float*)   carve((size_t)NB * 107 * 87 * 4);
    float*    a2  = (float*)   carve((size_t)NB * 102 * 82 * 4);
    float*    a3  = (float*)   carve((size_t)NB * 97  * 77 * 4);
    float*    a4  = (float*)   carve((size_t)NB * 92  * 72 * 4);
    _Float16* h16 = (_Float16*)carve((size_t)NB * FC1_KP * 2);
    _Float16* w1h = (_Float16*)carve((size_t)FC1_NP * FC1_KP * 2);
    _Float16* r1h = (_Float16*)carve((size_t)NB * FC2_KP * 2);
    _Float16* w2h = (_Float16*)carve((size_t)FC2_NP * FC2_KP * 2);
    (void)ws_size; (void)in_sizes; (void)n_in; (void)out_size;

    const int T = 256;
    auto blocks = [](int n, int t) { return (n + t - 1) / t; };

    // 0) zero padded f16 activation buffers (pad columns must read as 0 in WMMA)
    fill0_k<<<blocks(NB * FC1_KP, T), T, 0, stream>>>(h16, NB * FC1_KP);
    fill0_k<<<blocks(NB * FC2_KP, T), T, 0, stream>>>(r1h, NB * FC2_KP);

    // 1) convert + pad FC weights to f16
    cvtpad_k<<<blocks(FC1_NP * FC1_KP, T), T, 0, stream>>>(fc1_w, w1h, FC1_N, FC1_K, FC1_NP, FC1_KP);
    cvtpad_k<<<blocks(FC2_NP * FC2_KP, T), T, 0, stream>>>(fc2_w, w2h, FC2_N, FC2_K, FC2_NP, FC2_KP);

    // 2) locally-connected conv stack; per-block weight staging in LDS (64x reuse)
    conv1_staged_k<<<blocks(107 * 87, 64), T, 0, stream>>>(x, w1, b1, a1);
    convm_staged_k<<<blocks(102 * 82, 64), T, 0, stream>>>(a1, w2, b2, a2, 107, 87, 102, 82);
    convm_staged_k<<<blocks(97  * 77, 64), T, 0, stream>>>(a2, w3, b3, a3, 102, 82, 97, 77);
    convm_staged_k<<<blocks(92  * 72, 64), T, 0, stream>>>(a3, w4, b4, a4, 97, 77, 92, 72);
    pool_k<<<blocks(NB * 46 * 36, T), T, 0, stream>>>(a4, h16);

    // 3) FC layers via v_wmma_f32_16x16x32_f16 (one wave per 16x16 tile)
    {
        dim3 g1(FC1_NP / 16, NB / 16);  // 19 x 4 tiles
        wmma_fc_k<<<g1, 32, 0, stream>>>(h16, w1h, fc1_b, FC1_KP, FC1_N,
                                         (float*)nullptr, 0, r1h, FC2_KP, /*relu=*/1);
        dim3 g2(FC2_NP / 16, NB / 16);  // 13 x 4 tiles
        wmma_fc_k<<<g2, 32, 0, stream>>>(r1h, w2h, fc2_b, FC2_KP, FC2_N,
                                         out, FC2_N, (_Float16*)nullptr, 0, /*relu=*/0);
    }
}
